// PC_model_15333033246804
// MI455X (gfx1250) — compile-verified, matrix operand order checked by reading, not run on hardware
//
#include <hip/hip_runtime.h>
#include <stdint.h>

typedef _Float16 half_t;
typedef __attribute__((ext_vector_type(16))) _Float16 v16h;
typedef __attribute__((ext_vector_type(8)))  _Float16 v8h;
typedef __attribute__((ext_vector_type(8)))  float    v8f;

#define BB   4
#define NPTS 8192
#define RV   32
#define NVOX 32768

// ======================= WMMA GEMM (f16 in, f32 acc) =======================
// A: [M, K] f16 row-major, K % 32 == 0, M % 256 == 0.
// Wt: [CO, K] f16 row-major. Y: [M, CO] f32.
// 256 threads = 8 waves; each wave owns 32 rows (2 A-frags) x CO cols.
// Double-buffered LDS tiles filled by async DMA; s_wait_asynccnt NLD retires
// only the older tile so the newer tile's DMA overlaps the WMMAs.
template<int NT>
__global__ __launch_bounds__(256) void gemm_wmma(const half_t* __restrict__ A,
    const half_t* __restrict__ Wt, float* __restrict__ Y, int M, int K) {
  constexpr int CO  = NT * 16;
  constexpr int NB  = (CO * 4 + 255) / 256;  // B-tile chunks per thread (1 or 2)
  constexpr int NLD = 4 + NB;                // async loads per tile per thread/wave
  __shared__ half_t sA[2][256 * 32];
  __shared__ half_t sB[2][CO * 32];
  const int tid = threadIdx.x, lane = tid & 31, wave = tid >> 5;
  const int rowBase = blockIdx.x * 256;
  v8f acc[2][NT] = {};

  auto issueTile = [&](int k0, int buf) {
    // A tile: 256 rows x 32 halves = 1024 x 16B chunks, 4 per thread
#pragma unroll
    for (int it = 0; it < 4; ++it) {
      int ci = tid + it * 256;
      int r = ci >> 2, part = ci & 3;
      unsigned lds = (unsigned)(size_t)&sA[buf][r * 32 + part * 8];
      unsigned long long gp =
          (unsigned long long)(size_t)(A + (size_t)(rowBase + r) * K + k0 + part * 8);
      asm volatile("global_load_async_to_lds_b128 %0, %1, off" :: "v"(lds), "v"(gp) : "memory");
    }
    // B tile: CO rows x 32 halves; clamp so every thread issues exactly NB loads
#pragma unroll
    for (int it = 0; it < NB; ++it) {
      int ci = tid + it * 256;
      if (ci >= CO * 4) ci = CO * 4 - 1;
      int r = ci >> 2, part = ci & 3;
      unsigned lds = (unsigned)(size_t)&sB[buf][r * 32 + part * 8];
      unsigned long long gp =
          (unsigned long long)(size_t)(Wt + (size_t)r * K + k0 + part * 8);
      asm volatile("global_load_async_to_lds_b128 %0, %1, off" :: "v"(lds), "v"(gp) : "memory");
    }
  };

  issueTile(0, 0);
  const int nsteps = K / 32;
  for (int s = 0; s < nsteps; ++s) {
    int buf = s & 1;
    if (s + 1 < nsteps) {
      issueTile((s + 1) * 32, buf ^ 1);  // newest tile's DMA overlaps the wait + math
      asm volatile("s_wait_asynccnt %0" :: "n"(NLD) : "memory");  // retire older tile
    } else {
      asm volatile("s_wait_asynccnt 0x0" ::: "memory");
    }
    __syncthreads();
#pragma unroll
    for (int f = 0; f < 2; ++f) {
      int r = wave * 32 + f * 16 + (lane & 15);
      v16h a = *(const v16h*)&sA[buf][r * 32 + (lane >> 4) * 16];
#pragma unroll
      for (int t = 0; t < NT; ++t) {
        v16h bf = *(const v16h*)&sB[buf][(t * 16 + (lane & 15)) * 32 + (lane >> 4) * 16];
        acc[f][t] = __builtin_amdgcn_wmma_f32_16x16x32_f16(false, a, false, bf,
                                                           (short)0, acc[f][t], false, false);
      }
    }
    __syncthreads();
  }
  const int col = lane & 15, mo = (lane >> 4) * 8;
#pragma unroll
  for (int f = 0; f < 2; ++f)
#pragma unroll
    for (int t = 0; t < NT; ++t)
#pragma unroll
      for (int j = 0; j < 8; ++j)
        Y[(size_t)(rowBase + wave * 32 + f * 16 + mo + j) * CO + t * 16 + col] = acc[f][t][j];
}

// ================= Implicit-GEMM 3^3 conv, pad=1, via WMMA =================
// act: [B, D*H*W, Cin] f16 channel-last.  Wt: [CO, 27*Cin] f16 (k = off*Cin+ci).
// Y: [B, D*H*W, CO] f32.  grid = (nvox/128, B).  Cin % 32 == 0.
// Software-pipelined: next K-step's halo-checked chunks staged in registers
// while current step's WMMAs run; double LDS buffers -> 1 barrier per step.
template<int NT>
__global__ __launch_bounds__(256) void conv3_wmma(const half_t* __restrict__ act,
    const half_t* __restrict__ Wt, float* __restrict__ Y, int D, int H, int Wd, int Cin) {
  constexpr int CO = NT * 16;
  constexpr int NB = (CO * 4 + 255) / 256;
  __shared__ half_t sA[2][128 * 32];
  __shared__ half_t sB[2][CO * 32];
  const int tid = threadIdx.x, lane = tid & 31, wave = tid >> 5;
  const int nvox = D * H * Wd;
  const int b = blockIdx.y;
  const int voxBase = blockIdx.x * 128;
  const int cblks = Cin >> 5;
  const int Ksteps = 27 * cblks;
  const int K = 27 * Cin;
  v8f acc[NT] = {};
  v8h ra[2];
  v8h rb[NB];

  auto loadTile = [&](int ks) {
    int o = ks / cblks, cb0 = (ks % cblks) * 32;
    int kd = o / 9 - 1, kh = (o / 3) % 3 - 1, kw = o % 3 - 1;
#pragma unroll
    for (int it = 0; it < 2; ++it) {
      int ci = tid + it * 256;
      int r = ci >> 2, part = ci & 3;
      int v = voxBase + r;
      int d = v / (H * Wd), rem = v % (H * Wd), h = rem / Wd, w = rem % Wd;
      int dd = d + kd, hh = h + kh, ww = w + kw;
      v8h val = {};
      if (dd >= 0 && dd < D && hh >= 0 && hh < H && ww >= 0 && ww < Wd) {
        size_t gi = (((size_t)b * nvox) + (size_t)((dd * H + hh) * Wd + ww)) * Cin + cb0 + part * 8;
        val = *(const v8h*)&act[gi];
      }
      ra[it] = val;
    }
#pragma unroll
    for (int it = 0; it < NB; ++it) {
      int ci = tid + it * 256;
      if (ci >= CO * 4) ci = CO * 4 - 1;
      int r = ci >> 2, part = ci & 3;
      rb[it] = *(const v8h*)&Wt[(size_t)r * K + o * Cin + cb0 + part * 8];
    }
    if (ks + 1 < Ksteps)
      __builtin_prefetch(&Wt[(size_t)(tid & (CO - 1)) * K + ((ks + 1) / cblks) * Cin], 0, 0);
  };

  loadTile(0);
  for (int ks = 0; ks < Ksteps; ++ks) {
    int buf = ks & 1;
#pragma unroll
    for (int it = 0; it < 2; ++it) {
      int ci = tid + it * 256;
      int r = ci >> 2, part = ci & 3;
      *(v8h*)&sA[buf][r * 32 + part * 8] = ra[it];
    }
#pragma unroll
    for (int it = 0; it < NB; ++it) {
      int ci = tid + it * 256;
      if (ci >= CO * 4) ci = CO * 4 - 1;
      int r = ci >> 2, part = ci & 3;
      *(v8h*)&sB[buf][r * 32 + part * 8] = rb[it];
    }
    __syncthreads();
    if (ks + 1 < Ksteps) loadTile(ks + 1);  // global latency hides under WMMAs
    {
      int r = wave * 16 + (lane & 15);
      v16h a = *(const v16h*)&sA[buf][r * 32 + (lane >> 4) * 16];
#pragma unroll
      for (int t = 0; t < NT; ++t) {
        v16h bf = *(const v16h*)&sB[buf][(t * 16 + (lane & 15)) * 32 + (lane >> 4) * 16];
        acc[t] = __builtin_amdgcn_wmma_f32_16x16x32_f16(false, a, false, bf,
                                                        (short)0, acc[t], false, false);
      }
    }
    // no trailing barrier: next store targets the opposite LDS buffer
  }
  const int col = lane & 15, mo = (lane >> 4) * 8;
#pragma unroll
  for (int t = 0; t < NT; ++t)
#pragma unroll
    for (int j = 0; j < 8; ++j)
      Y[((size_t)b * nvox + voxBase + wave * 16 + mo + j) * CO + t * 16 + col] = acc[t][j];
}

// ======================= weight packing to f16 =======================
__global__ void wcvt_lin(const float* w, half_t* o, int CO, int CI, int Kp) {
  int t = blockIdx.x * 256 + threadIdx.x;
  if (t >= CO * Kp) return;
  int c = t / Kp, k = t % Kp;
  o[t] = (half_t)(k < CI ? w[(size_t)c * CI + k] : 0.f);
}
__global__ void wcvt_c3(const float* w, half_t* o, int CO, int CI) {
  int K = 27 * CI;
  int t = blockIdx.x * 256 + threadIdx.x;
  if (t >= CO * K) return;
  int c = t / K, k = t % K;
  int off = k / CI, ci = k % CI;
  o[t] = (half_t)w[((size_t)c * CI + ci) * 27 + off];
}

// ======================= point-cloud front end =======================
__global__ void prep_points(const float* fd, float* pts, int* valid, int total, int N) {
  int t = blockIdx.x * 256 + threadIdx.x;
  if (t >= total) return;
  int b = t / N;
  const float* p = fd + (size_t)t * 4;
  pts[(size_t)t * 3 + 0] = p[0] / 32767.f;
  pts[(size_t)t * 3 + 1] = p[1] / 32767.f;
  pts[(size_t)t * 3 + 2] = p[2] / 32767.f;
  if (p[3] > 0.5f) atomicAdd(&valid[b], 1);
}
__global__ void gather_xyz(const float* pts, const int* valid, float* xyz, float* lf0,
                           int total, int N) {
  int t = blockIdx.x * 256 + threadIdx.x;
  if (t >= total) return;
  int b = t / N, n = t % N;
  int M = 1;
  for (int i = 0; i < BB; ++i) M = (valid[i] > M) ? valid[i] : M;
  int vb = valid[b] > 0 ? valid[b] : 1;
  int id = (n % M) % vb;
  xyz[(size_t)t * 3 + 0] = pts[((size_t)b * N + id) * 3 + 0];
  xyz[(size_t)t * 3 + 1] = pts[((size_t)b * N + id) * 3 + 1];
  xyz[(size_t)t * 3 + 2] = pts[((size_t)b * N + id) * 3 + 2];
  lf0[((size_t)b * 3 + 0) * N + n] = pts[(size_t)t * 3 + 0];
  lf0[((size_t)b * 3 + 1) * N + n] = pts[(size_t)t * 3 + 1];
  lf0[((size_t)b * 3 + 2) * N + n] = pts[(size_t)t * 3 + 2];
}
__global__ void fps_kernel(const float* xyz, int Np, int S, int* outIdx, float* dist) {
  int b = blockIdx.x;
  const float* X = xyz + (size_t)b * Np * 3;
  float* Dd = dist + (size_t)b * Np;
  int* O = outIdx + (size_t)b * S;
  __shared__ float sv[256];
  __shared__ int si[256];
  for (int i = threadIdx.x; i < Np; i += 256) Dd[i] = 1e10f;
  __syncthreads();
  int last = 0;
  for (int s = 0; s < S; ++s) {
    if (threadIdx.x == 0) O[s] = last;
    float lx = X[last * 3], ly = X[last * 3 + 1], lz = X[last * 3 + 2];
    float bv = -1.f; int bi = 0;
    for (int i = threadIdx.x; i < Np; i += 256) {
      float dx = X[i * 3] - lx, dy = X[i * 3 + 1] - ly, dz = X[i * 3 + 2] - lz;
      float d = dx * dx + dy * dy + dz * dz;
      float cur = Dd[i];
      if (d < cur) { cur = d; Dd[i] = cur; }
      if (cur > bv) { bv = cur; bi = i; }
    }
    sv[threadIdx.x] = bv; si[threadIdx.x] = bi;
    __syncthreads();
    for (int o = 128; o > 0; o >>= 1) {
      if (threadIdx.x < o && sv[threadIdx.x + o] > sv[threadIdx.x]) {
        sv[threadIdx.x] = sv[threadIdx.x + o]; si[threadIdx.x] = si[threadIdx.x + o];
      }
      __syncthreads();
    }
    last = si[0];
    __syncthreads();
  }
}
__global__ void gather_centers(const float* xyz, int Np, const int* fidx, float* nxyz,
                               int total, int S) {
  int t = blockIdx.x * 256 + threadIdx.x;
  if (t >= total) return;
  int b = t / S;
  int id = fidx[t];
  nxyz[(size_t)t * 3 + 0] = xyz[((size_t)b * Np + id) * 3 + 0];
  nxyz[(size_t)t * 3 + 1] = xyz[((size_t)b * Np + id) * 3 + 1];
  nxyz[(size_t)t * 3 + 2] = xyz[((size_t)b * Np + id) * 3 + 2];
}
__global__ void ballq(const float* xyz, int Np, const float* nxyz, float r2, int ns,
                      int* idx, int total, int S) {
  int t = blockIdx.x * 256 + threadIdx.x;
  if (t >= total) return;
  int b = t / S;
  float cx = nxyz[(size_t)t * 3], cy = nxyz[(size_t)t * 3 + 1], cz = nxyz[(size_t)t * 3 + 2];
  int cnt = 0, first = -1;
  for (int i = 0; i < Np && cnt < ns; ++i) {
    float dx = xyz[((size_t)b * Np + i) * 3 + 0] - cx;
    float dy = xyz[((size_t)b * Np + i) * 3 + 1] - cy;
    float dz = xyz[((size_t)b * Np + i) * 3 + 2] - cz;
    if (dx * dx + dy * dy + dz * dz < r2) {
      if (first < 0) first = i;
      idx[(size_t)t * ns + cnt] = i;
      cnt++;
    }
  }
  if (first < 0) first = 0;
  for (; cnt < ns; ++cnt) idx[(size_t)t * ns + cnt] = first;
}
// Build grouped SA input X[M, Kp]: [g_xyz(3) | feats(Cf) | 0-pad]
__global__ void sa_group(const float* xyz, int Np, const float* nxyz, int S, const int* idx,
                         int ns, const float* feats, int Cf, int Nf, half_t* X, int Kp, int total) {
  int t = blockIdx.x * 256 + threadIdx.x;
  if (t >= total) return;
  int m = t / Kp, k = t % Kp;
  int b = m / (S * ns), rem = m % (S * ns), s = rem / ns, n = rem % ns;
  int id = idx[((size_t)b * S + s) * ns + n];
  if (id >= Np) id = Np - 1;
  float v = 0.f;
  if (k < 3)
    v = xyz[((size_t)b * Np + id) * 3 + k] - nxyz[((size_t)b * S + s) * 3 + k];
  else if (k < 3 + Cf)
    v = feats[((size_t)b * Cf + (k - 3)) * Nf + id];
  X[t] = (half_t)v;
}

// ======================= stats + fused apply =======================
__global__ void bstats(const float* Y, int M, int C, float* mean, float* rsig) {
  int c = blockIdx.x;
  float s = 0.f, ss = 0.f;
  for (int m = threadIdx.x; m < M; m += 256) {
    float v = Y[(size_t)m * C + c];
    s += v; ss += v * v;
  }
  __shared__ float s1[256], s2[256];
  s1[threadIdx.x] = s; s2[threadIdx.x] = ss;
  __syncthreads();
  for (int o = 128; o > 0; o >>= 1) {
    if (threadIdx.x < o) { s1[threadIdx.x] += s1[threadIdx.x + o]; s2[threadIdx.x] += s2[threadIdx.x + o]; }
    __syncthreads();
  }
  if (threadIdx.x == 0) {
    float mu = s1[0] / M, var = s2[0] / M - mu * mu;
    mean[c] = mu; rsig[c] = rsqrtf(var + 1e-5f);
  }
}
__global__ void istats(const float* Y, int nv, int C, float* mean, float* rsig) {
  int b = blockIdx.x / C, c = blockIdx.x % C;
  float s = 0.f, ss = 0.f;
  for (int v = threadIdx.x; v < nv; v += 256) {
    float x = Y[((size_t)b * nv + v) * C + c];
    s += x; ss += x * x;
  }
  __shared__ float s1[256], s2[256];
  s1[threadIdx.x] = s; s2[threadIdx.x] = ss;
  __syncthreads();
  for (int o = 128; o > 0; o >>= 1) {
    if (threadIdx.x < o) { s1[threadIdx.x] += s1[threadIdx.x + o]; s2[threadIdx.x] += s2[threadIdx.x + o]; }
    __syncthreads();
  }
  if (threadIdx.x == 0) {
    float mu = s1[0] / nv, var = s2[0] / nv - mu * mu;
    mean[blockIdx.x] = mu; rsig[blockIdx.x] = rsqrtf(var + 1e-5f);
  }
}
// mode 0: BN affine + relu; 1: inorm + relu; 2: bias + relu; 3: bias only.
// optional residual add (post-relu) and dual f16/f32 outputs.
__global__ void apply_act(const float* Y, const float* mean, const float* rsig,
                          const float* g, const float* bta, const float* bias, const float* res,
                          half_t* o16, float* o32, int M, int C, int mode, int nvB) {
  int t = blockIdx.x * 256 + threadIdx.x;
  if (t >= M * C) return;
  int m = t / C, c = t % C;
  float v = Y[t];
  if (mode == 0) v = (v - mean[c]) * rsig[c] * g[c] + bta[c];
  else if (mode == 1) { int s = (m / nvB) * C + c; v = (v - mean[s]) * rsig[s]; }
  else v += bias[c];
  if (mode != 3) v = fmaxf(v, 0.f);
  if (res) v += res[t];
  if (o16) o16[t] = (half_t)v;
  if (o32) o32[t] = v;
}
__global__ void sa_apply(const float* Y, const float* mean, const float* rsig,
                         const float* g, const float* bta, int S, int ns, int CO,
                         float* out, int Ctot, int cofs, int total) {
  int t = blockIdx.x * 256 + threadIdx.x;
  if (t >= total) return;
  int s = t % S, o = (t / S) % CO, b = t / (S * CO);
  float mu = mean[o], rs = rsig[o], gg = g[o], bb = bta[o];
  float mx = 0.f;
  for (int n = 0; n < ns; ++n) {
    float v = Y[((size_t)(((size_t)b * S + s) * ns + n)) * CO + o];
    v = fmaxf((v - mu) * rs * gg + bb, 0.f);
    mx = fmaxf(mx, v);
  }
  out[((size_t)b * Ctot + cofs + o) * S + s] = mx;
}
__global__ void fp_apply(const float* Y, const float* mean, const float* rsig,
                         const float* g, const float* bta, float* out, int Np, int CO, int total) {
  int t = blockIdx.x * 256 + threadIdx.x;
  if (t >= total) return;
  int n = t % Np, o = (t / Np) % CO, b = t / (Np * CO);
  float v = Y[((size_t)b * Np + n) * CO + o];
  v = (v - mean[o]) * rsig[o] * g[o] + bta[o];
  out[((size_t)b * CO + o) * Np + n] = fmaxf(v, 0.f);
}

// ======================= FP interpolation =======================
__global__ void fp_build(const float* x1, int N1, const float* x2, int N2,
                         const float* f1, int C1, const float* f2, int C2,
                         half_t* X, int Kp, int total) {
  int t = blockIdx.x * 256 + threadIdx.x;
  if (t >= total) return;
  int b = t / N1, n = t % N1;
  float px = x1[(size_t)t * 3], py = x1[(size_t)t * 3 + 1], pz = x1[(size_t)t * 3 + 2];
  float d0 = 1e30f, d1 = 1e30f, d2 = 1e30f;
  int i0 = 0, i1 = 0, i2 = 0;
  for (int j = 0; j < N2; ++j) {
    const float* q = &x2[((size_t)b * N2 + j) * 3];
    float dx = px - q[0], dy = py - q[1], dz = pz - q[2];
    float d = dx * dx + dy * dy + dz * dz;
    if (d < d0) { d2 = d1; i2 = i1; d1 = d0; i1 = i0; d0 = d; i0 = j; }
    else if (d < d1) { d2 = d1; i2 = i1; d1 = d; i1 = j; }
    else if (d < d2) { d2 = d; i2 = j; }
  }
  float w0 = 1.f / (d0 + 1e-8f), w1 = 1.f / (d1 + 1e-8f), w2 = 1.f / (d2 + 1e-8f);
  float sw = w0 + w1 + w2;
  w0 /= sw; w1 /= sw; w2 /= sw;
  half_t* row = X + (size_t)t * Kp;
  for (int c = 0; c < C2; ++c) {
    const float* fb = &f2[((size_t)b * C2 + c) * N2];
    row[c] = (half_t)(w0 * fb[i0] + w1 * fb[i1] + w2 * fb[i2]);
  }
  for (int c = 0; c < C1; ++c) row[C2 + c] = (half_t)f1[((size_t)b * C1 + c) * N1 + n];
  for (int k = C2 + C1; k < Kp; ++k) row[k] = (half_t)0.f;
}
__global__ void chlast_cvt(const float* in, int C, int Np, half_t* X, int Kp, int total) {
  int t = blockIdx.x * 256 + threadIdx.x;
  if (t >= total) return;
  int k = t % Kp, m = t / Kp, b = m / Np, n = m % Np;
  X[t] = (half_t)(k < C ? in[((size_t)b * C + k) * Np + n] : 0.f);
}

// ======================= voxelize =======================
__global__ void voxel_acc(const float* pts, const float* h, float* vs, float* vc,
                          int total, int N) {
  int t = blockIdx.x * 256 + threadIdx.x;
  if (t >= total) return;
  int b = t / N;
  int vi[3];
  for (int k = 0; k < 3; ++k) {
    float f = (pts[(size_t)t * 3 + k] + 1.f) * 0.5f * RV;
    f = fminf(fmaxf(f, 0.f), (float)(RV - 1));
    vi[k] = (int)(f + 0.5f);
    if (vi[k] > RV - 1) vi[k] = RV - 1;
  }
  int flat = (vi[0] * RV + vi[1]) * RV + vi[2];
  for (int c = 0; c < 32; ++c)
    atomicAdd(&vs[((size_t)b * NVOX + flat) * 32 + c], h[(size_t)t * 32 + c]);
  atomicAdd(&vc[(size_t)b * NVOX + flat], 1.f);
}
__global__ void voxel_div(const float* vs, const float* vc, half_t* o, int total) {
  int t = blockIdx.x * 256 + threadIdx.x;
  if (t >= total) return;
  int v = (t / 32) % NVOX, b = t / (32 * NVOX);
  o[t] = (half_t)(vs[t] / fmaxf(vc[(size_t)b * NVOX + v], 1.f));
}

// ======================= unet glue =======================
__global__ void maxpool2k(const float* in, int D, int H, int W, int C,
                          half_t* o16, float* o32, int total) {
  int t = blockIdx.x * 256 + threadIdx.x;
  if (t >= total) return;
  int c = t % C, v = t / C;
  int oW = W / 2, oH = H / 2, oD = D / 2;
  int ow = v % oW; v /= oW;
  int oh = v % oH; v /= oH;
  int od = v % oD; int b = v / oD;
  float m = -1e30f;
  for (int dz = 0; dz < 2; ++dz)
    for (int dy = 0; dy < 2; ++dy)
      for (int dx = 0; dx < 2; ++dx) {
        size_t gi = ((((size_t)b * D + od * 2 + dz) * H + oh * 2 + dy) * W + ow * 2 + dx) * C + c;
        m = fmaxf(m, in[gi]);
      }
  o16[t] = (half_t)m;
  o32[t] = m;
}
__global__ void up2k(const half_t* in, int D, int H, int W, int C, half_t* o, int total) {
  int t = blockIdx.x * 256 + threadIdx.x;
  if (t >= total) return;
  int c = t % C, v = t / C;
  int oW = W * 2, oH = H * 2, oD = D * 2;
  int ow = v % oW; v /= oW;
  int oh = v % oH; v /= oH;
  int od = v % oD; int b = v / oD;
  o[t] = in[((((size_t)b * D + od / 2) * H + oh / 2) * W + ow / 2) * C + c];
}
__global__ void out_conv(const float* u, const float* w, const float* bias, float* o, int total) {
  int t = blockIdx.x * 256 + threadIdx.x;
  if (t >= total) return;
  float s = bias[0];
  for (int c = 0; c < 32; ++c) s += u[(size_t)t * 32 + c] * w[c];
  o[t] = s;
}

// =============================== host ===============================
extern "C" void kernel_launch(void* const* d_in, const int* in_sizes, int n_in,
                              void* d_out, int out_size, void* d_ws, size_t ws_size,
                              hipStream_t stream) {
  (void)in_sizes; (void)n_in; (void)out_size; (void)ws_size;
  auto F = [&](int i) { return (const float*)d_in[i]; };
  auto K1 = [](long n) { return (int)((n + 255) / 256); };

  char* wsp = (char*)d_ws;
  size_t off = 0;
  auto alloc = [&](size_t bytes) -> void* {
    off = (off + 255) & ~(size_t)255;
    void* p = wsp + off;
    off += bytes;
    return p;
  };
  const int Ns[5] = {NPTS, 1024, 256, 64, 16};
  const int lfC[5] = {3, 48, 96, 192, 256};

  float* pts  = (float*)alloc((size_t)BB * NPTS * 3 * 4);
  float* xyzg = (float*)alloc((size_t)BB * NPTS * 3 * 4);
  float* lf0  = (float*)alloc((size_t)BB * 3 * NPTS * 4);
  int* valid  = (int*)alloc(64);
  int* fpsidx = (int*)alloc((size_t)BB * 1024 * 4);
  float* distb = (float*)alloc((size_t)BB * NPTS * 4);
  int* ballidx = (int*)alloc((size_t)BB * 1024 * 32 * 4);
  float* lxyz[5]; lxyz[0] = xyzg;
  for (int l = 1; l <= 4; ++l) lxyz[l] = (float*)alloc((size_t)BB * Ns[l] * 3 * 4);
  float* lf[5]; lf[0] = lf0;
  for (int l = 1; l <= 4; ++l) lf[l] = (float*)alloc((size_t)BB * lfC[l] * Ns[l] * 4);
  const int fpoC[4] = {32, 32, 64, 128};
  float* fpo[4];
  for (int k = 0; k < 4; ++k) fpo[k] = (float*)alloc((size_t)BB * fpoC[k] * Ns[k] * 4);
  float* meanb = (float*)alloc(1024 * 4);
  float* rsigb = (float*)alloc(1024 * 4);
  half_t* Xbuf  = (half_t*)alloc((size_t)8 << 20);
  half_t* X2buf = (half_t*)alloc((size_t)2 << 20);
  half_t* Wc    = (half_t*)alloc((size_t)1 << 20);
  float* Ybuf   = (float*)alloc((size_t)32 << 20);
  float* h32    = (float*)alloc((size_t)BB * NPTS * 32 * 4);
  float* voxsum = (float*)alloc((size_t)BB * NVOX * 32 * 4);
  float* voxcnt = (float*)alloc((size_t)BB * NVOX * 4);
  half_t* vox16 = (half_t*)alloc((size_t)BB * NVOX * 32 * 2);
  const int xsD[4] = {32, 16, 8, 4};
  const int xsC[4] = {32, 64, 128, 128};
  float* xs32[4]; half_t* xs16[4];
  for (int i = 0; i < 4; ++i) {
    size_t nv = (size_t)xsD[i] * xsD[i] * xsD[i];
    xs32[i] = (float*)alloc((size_t)BB * nv * xsC[i] * 4);
    xs16[i] = (half_t*)alloc((size_t)BB * nv * xsC[i] * 2);
  }
  float* P0_32 = (float*)alloc((size_t)32 << 20);
  half_t* P0_16 = (half_t*)alloc((size_t)16 << 20);
  float* P1_32 = (float*)alloc((size_t)32 << 20);
  half_t* P2_16 = (half_t*)alloc((size_t)16 << 20);
  float* IT_32 = (float*)alloc((size_t)16 << 20);
  half_t* IT_16 = (half_t*)alloc((size_t)8 << 20);
  float* U_32 = (float*)alloc((size_t)16 << 20);
  half_t* U_16 = (half_t*)alloc((size_t)8 << 20);

  auto gemmL = [&](const half_t* A, const half_t* W, float* Yp, int M, int K, int CO) {
    int g = M / 256;
    if (CO == 16)      gemm_wmma<1><<<g, 256, 0, stream>>>(A, W, Yp, M, K);
    else if (CO == 32) gemm_wmma<2><<<g, 256, 0, stream>>>(A, W, Yp, M, K);
    else if (CO == 64) gemm_wmma<4><<<g, 256, 0, stream>>>(A, W, Yp, M, K);
    else               gemm_wmma<8><<<g, 256, 0, stream>>>(A, W, Yp, M, K);
  };
  auto convL = [&](const half_t* A, const half_t* W, float* Yp, int D, int CO, int CI) {
    int nv = D * D * D;
    dim3 g(nv / 128, BB);
    if (CO == 32)      conv3_wmma<2><<<g, 256, 0, stream>>>(A, W, Yp, D, D, D, CI);
    else if (CO == 64) conv3_wmma<4><<<g, 256, 0, stream>>>(A, W, Yp, D, D, D, CI);
    else               conv3_wmma<8><<<g, 256, 0, stream>>>(A, W, Yp, D, D, D, CI);
  };

  // ---- front end ----
  hipMemsetAsync(valid, 0, 64, stream);
  prep_points<<<K1((long)BB * NPTS), 256, 0, stream>>>(F(0), pts, valid, BB * NPTS, NPTS);
  gather_xyz<<<K1((long)BB * NPTS), 256, 0, stream>>>(pts, valid, xyzg, lf0, BB * NPTS, NPTS);

  // ---- SA (set abstraction, MSG) ----
  const float RAD[4][2] = {{0.05f, 0.1f}, {0.1f, 0.2f}, {0.2f, 0.4f}, {0.4f, 0.8f}};
  const int NSA[2] = {16, 32};
  const int SACO[4][2] = {{16, 32}, {32, 64}, {64, 128}, {128, 128}};
  const int KP[4] = {32, 64, 128, 224};
  for (int lvl = 0; lvl < 4; ++lvl) {
    int S = Ns[lvl + 1], Np = Ns[lvl], Cf = lfC[lvl], Ctot = lfC[lvl + 1];
    fps_kernel<<<BB, 256, 0, stream>>>(lxyz[lvl], Np, S, fpsidx, distb);
    gather_centers<<<K1((long)BB * S), 256, 0, stream>>>(lxyz[lvl], Np, fpsidx, lxyz[lvl + 1], BB * S, S);
    int cofs = 0;
    for (int br = 0; br < 2; ++br) {
      int ns = NSA[br], CO = SACO[lvl][br], Kp = KP[lvl];
      float r = RAD[lvl][br];
      ballq<<<K1((long)BB * S), 256, 0, stream>>>(lxyz[lvl], Np, lxyz[lvl + 1], r * r, ns, ballidx, BB * S, S);
      int M = BB * S * ns;
      sa_group<<<K1((long)M * Kp), 256, 0, stream>>>(lxyz[lvl], Np, lxyz[lvl + 1], S, ballidx,
                                                     ns, lf[lvl], Cf, Np, Xbuf, Kp, M * Kp);
      int iw = 1 + (lvl * 2 + br) * 3;
      wcvt_lin<<<K1((long)CO * Kp), 256, 0, stream>>>(F(iw), Wc, CO, 3 + Cf, Kp);
      gemmL(Xbuf, Wc, Ybuf, M, Kp, CO);
      bstats<<<CO, 256, 0, stream>>>(Ybuf, M, CO, meanb, rsigb);
      sa_apply<<<K1((long)BB * CO * S), 256, 0, stream>>>(Ybuf, meanb, rsigb, F(iw + 1), F(iw + 2),
                                                          S, ns, CO, lf[lvl + 1], Ctot, cofs, BB * CO * S);
      cofs += CO;
    }
  }

  // ---- FP (feature propagation), applied coarse -> fine ----
  struct FPS_ { int lvl, j, C1, CO, Kp; const float* f2; int C2; };
  const FPS_ fpseq[4] = {
    {3, 3, 192, 128, 448, lf[4], 256},
    {2, 2, 96, 64, 224, fpo[3], 128},
    {1, 1, 48, 32, 128, fpo[2], 64},
    {0, 0, 3, 32, 64, fpo[1], 32},
  };
  for (int q = 0; q < 4; ++q) {
    const FPS_& e = fpseq[q];
    int N1 = Ns[e.lvl], N2 = Ns[e.lvl + 1];
    fp_build<<<K1((long)BB * N1), 256, 0, stream>>>(lxyz[e.lvl], N1, lxyz[e.lvl + 1], N2,
                                                    lf[e.lvl], e.C1, e.f2, e.C2, Xbuf, e.Kp, BB * N1);
    int iw = 25 + e.j * 3;
    wcvt_lin<<<K1((long)e.CO * e.Kp), 256, 0, stream>>>(F(iw), Wc, e.CO, e.C2 + e.C1, e.Kp);
    int M = BB * N1;
    gemmL(Xbuf, Wc, Ybuf, M, e.Kp, e.CO);
    bstats<<<e.CO, 256, 0, stream>>>(Ybuf, M, e.CO, meanb, rsigb);
    fp_apply<<<K1((long)BB * e.CO * N1), 256, 0, stream>>>(Ybuf, meanb, rsigb, F(iw + 1), F(iw + 2),
                                                           fpo[e.lvl], N1, e.CO, BB * e.CO * N1);
  }

  // ---- FC head ----
  chlast_cvt<<<K1((long)BB * NPTS * 32), 256, 0, stream>>>(fpo[0], 32, NPTS, Xbuf, 32, BB * NPTS * 32);
  wcvt_lin<<<K1(32 * 32), 256, 0, stream>>>(F(37), Wc, 32, 32, 32);
  gemmL(Xbuf, Wc, Ybuf, BB * NPTS, 32, 32);
  bstats<<<32, 256, 0, stream>>>(Ybuf, BB * NPTS, 32, meanb, rsigb);
  apply_act<<<K1((long)BB * NPTS * 32), 256, 0, stream>>>(Ybuf, meanb, rsigb, F(38), F(39),
      nullptr, nullptr, X2buf, nullptr, BB * NPTS, 32, 0, 0);
  wcvt_lin<<<K1(32 * 32), 256, 0, stream>>>(F(40), Wc, 32, 32, 32);
  gemmL(X2buf, Wc, Ybuf, BB * NPTS, 32, 32);
  apply_act<<<K1((long)BB * NPTS * 32), 256, 0, stream>>>(Ybuf, nullptr, nullptr, nullptr, nullptr,
      F(41), nullptr, nullptr, h32, BB * NPTS, 32, 3, 0);

  // ---- voxelize ----
  hipMemsetAsync(voxsum, 0, (size_t)BB * NVOX * 32 * 4, stream);
  hipMemsetAsync(voxcnt, 0, (size_t)BB * NVOX * 4, stream);
  voxel_acc<<<K1((long)BB * NPTS), 256, 0, stream>>>(pts, h32, voxsum, voxcnt, BB * NPTS, NPTS);
  voxel_div<<<K1((long)BB * NVOX * 32), 256, 0, stream>>>(voxsum, voxcnt, vox16, BB * NVOX * 32);

  // ---- U-Net ----
  // c1: 1x1x1 conv (pure GEMM) + batch norm + relu
  wcvt_lin<<<K1(32 * 32), 256, 0, stream>>>(F(42), Wc, 32, 32, 32);
  gemmL(vox16, Wc, Ybuf, BB * NVOX, 32, 32);
  bstats<<<32, 256, 0, stream>>>(Ybuf, BB * NVOX, 32, meanb, rsigb);
  apply_act<<<K1((long)BB * NVOX * 32), 256, 0, stream>>>(Ybuf, meanb, rsigb, F(44), F(45),
      nullptr, nullptr, xs16[0], xs32[0], BB * NVOX, 32, 0, 0);

  auto convblock = [&](const half_t* in16, int D, int Cin, int Cout,
                       int wi1, int wi2, half_t* out16, float* out32) {
    int nv = D * D * D, M = BB * nv;
    wcvt_c3<<<K1((long)Cout * 27 * Cin), 256, 0, stream>>>(F(wi1), Wc, Cout, Cin);
    convL(in16, Wc, Ybuf, D, Cout, Cin);
    istats<<<BB * Cout, 256, 0, stream>>>(Ybuf, nv, Cout, meanb, rsigb);
    apply_act<<<K1((long)M * Cout), 256, 0, stream>>>(Ybuf, meanb, rsigb, nullptr, nullptr,
        nullptr, nullptr, P0_16, P0_32, M, Cout, 1, nv);
    wcvt_c3<<<K1((long)Cout * 27 * Cout), 256, 0, stream>>>(F(wi2), Wc, Cout, Cout);
    convL(P0_16, Wc, Ybuf, D, Cout, Cout);
    istats<<<BB * Cout, 256, 0, stream>>>(Ybuf, nv, Cout, meanb, rsigb);
    apply_act<<<K1((long)M * Cout), 256, 0, stream>>>(Ybuf, meanb, rsigb, nullptr, nullptr,
        nullptr, P0_32, out16, out32, M, Cout, 1, nv);
  };

  const int encD[3] = {32, 16, 8};
  const int encCi[3] = {32, 64, 128};
  const int encCo[3] = {64, 128, 128};
  for (int i = 0; i < 3; ++i) {
    convblock(xs16[i], encD[i], encCi[i], encCo[i], 46 + i * 4, 48 + i * 4, nullptr, P1_32);
    int oD = encD[i] / 2;
    long tot = (long)BB * oD * oD * oD * encCo[i];
    maxpool2k<<<K1(tot), 256, 0, stream>>>(P1_32, encD[i], encD[i], encD[i], encCo[i],
                                           xs16[i + 1], xs32[i + 1], (int)tot);
  }

  const int upCO[3] = {32, 64, 128};
  const half_t* u16 = xs16[3];
  int uD = 4, uC = 128;
  for (int i = 2; i >= 0; --i) {
    int Db = encD[i];
    long upTot = (long)BB * Db * Db * Db * uC;
    up2k<<<K1(upTot), 256, 0, stream>>>(u16, uD, uD, uD, uC, P2_16, (int)upTot);
    wcvt_c3<<<K1((long)upCO[i] * 27 * uC), 256, 0, stream>>>(F(58 + i * 2), Wc, upCO[i], uC);
    convL(P2_16, Wc, Ybuf, Db, upCO[i], uC);
    int nv = Db * Db * Db, M = BB * nv;
    apply_act<<<K1((long)M * upCO[i]), 256, 0, stream>>>(Ybuf, nullptr, nullptr, nullptr, nullptr,
        F(59 + i * 2), xs32[i], IT_16, IT_32, M, upCO[i], 2, 0);
    convblock(IT_16, Db, upCO[i], upCO[i], 64 + i * 4, 66 + i * 4, U_16, U_32);
    u16 = U_16; uD = Db; uC = upCO[i];
  }

  out_conv<<<K1((long)BB * NVOX), 256, 0, stream>>>(U_32, F(76), F(77), (float*)d_out, BB * NVOX);
}
